// StagewiseTNRD_28741921144854
// MI455X (gfx1250) — compile-verified, matrix operand order checked by reading, not run on hardware
//
#include <hip/hip_runtime.h>

typedef __attribute__((ext_vector_type(16))) _Float16 v16h;
typedef __attribute__((ext_vector_type(8)))  float    v8f;

#define EPSV 1e-6f
#define IMH 512
#define IMW 512
#define NSTAGES 7
#define NF 8
#define TILE 32

__device__ __forceinline__ int reflect_idx(int i, int n) {
    // jnp.pad(mode='reflect'): -1 -> 1, n -> n-2
    if (i < 0) i = -i;
    if (i >= n) i = 2 * (n - 1) - i;
    return i;
}

// CDNA5 has a hardware v_tanh_f32 (TRANS op -> co-executes with WMMA).
__device__ __forceinline__ float fast_tanh(float x) {
#if __has_builtin(__builtin_amdgcn_tanhf)
    return __builtin_amdgcn_tanhf(x);
#else
    // branchless fallback: tanh(x) = 1 - 2/(e^{2x}+1); overflow -> +/-1 naturally
    float e = __expf(2.0f * x);
    return 1.0f - 2.0f / (e + 1.0f);
#endif
}

// Async global->LDS copy of one dword per lane (gfx1250 ASYNCcnt path).
__device__ __forceinline__ void async_g2lds_b32(const float* gptr, float* lptr) {
    unsigned int loff = (unsigned int)(unsigned long long)lptr;  // generic shared addr[31:0] = LDS offset
    asm volatile("global_load_async_to_lds_b32 %0, %1, off"
                 :: "v"(loff), "v"(gptr) : "memory");
}
__device__ __forceinline__ void wait_async() {
    asm volatile("s_wait_asynccnt 0x0" ::: "memory");
}

__global__ void __launch_bounds__(256) init_umax_kernel(unsigned int* umax) {
    if (threadIdx.x < NSTAGES * 8) umax[threadIdx.x] = 0u;
}

// Stage part 1: 5x5 Gaussian (reflect pad) -> u_sigma; per-image max via atomicMax on bits.
__global__ void __launch_bounds__(256) gauss_max_kernel(
    const float* __restrict__ est, const float* __restrict__ gauss,
    float* __restrict__ usig, unsigned int* __restrict__ umax)
{
    __shared__ float sEst[36 * 36];
    __shared__ float sG[25];
    __shared__ float sMax[256];
    const int tid = threadIdx.x;
    const int b   = blockIdx.z;
    const int ty0 = blockIdx.y * TILE, tx0 = blockIdx.x * TILE;
    const float* img = est + (size_t)b * IMH * IMW;

    if (tid < 25) sG[tid] = gauss[tid];
    for (int i = tid; i < 36 * 36; i += 256) {
        int ly = i / 36, lx = i % 36;
        int gy = reflect_idx(ty0 + ly - 2, IMH);
        int gx = reflect_idx(tx0 + lx - 2, IMW);
        async_g2lds_b32(img + gy * IMW + gx, &sEst[i]);
    }
    wait_async();
    __syncthreads();

    float lmax = 0.f;
    float* uimg = usig + (size_t)b * IMH * IMW;
    for (int i = tid; i < TILE * TILE; i += 256) {
        int oy = i / TILE, ox = i % TILE;
        float acc = 0.f;
#pragma unroll
        for (int ky = 0; ky < 5; ++ky)
#pragma unroll
            for (int kx = 0; kx < 5; ++kx)
                acc = fmaf(sEst[(oy + ky) * 36 + ox + kx], sG[ky * 5 + kx], acc);
        uimg[(size_t)(ty0 + oy) * IMW + tx0 + ox] = acc;
        lmax = fmaxf(lmax, acc);
    }
    sMax[tid] = lmax;
    __syncthreads();
#pragma unroll
    for (int s = 128; s > 0; s >>= 1) {
        if (tid < s) sMax[tid] = fmaxf(sMax[tid], sMax[tid + s]);
        __syncthreads();
    }
    if (tid == 0) atomicMax(umax + b, __float_as_uint(sMax[0]));
}

// Stage part 2: fused  resp(WMMA) -> tanh/iw -> tmp(WMMA) -> diff gather -> fidelity/gate/clip
__global__ void __launch_bounds__(256) stage_update_kernel(
    const float* __restrict__ est_in, const float* __restrict__ noisy,
    const float* __restrict__ usig, const unsigned int* __restrict__ umax,
    const float* __restrict__ filt,              // [8][9] this stage
    const float* __restrict__ wv, const float* __restrict__ bv,   // [8] each
    const float* __restrict__ lamp, const float* __restrict__ gatep,
    float* __restrict__ est_out)
{
    __shared__ float sEst[36 * 36];        // est tile, halo 2, reflect
    __shared__ float sIw[34 * 34];         // influence weights, halo 1
    __shared__ float sTmp[9 * 34 * 36];    // tmp[t][py][px], row stride 36

    const int tid  = threadIdx.x;
    const int lane = tid & 31;
    const int wave = tid >> 5;
    const int b    = blockIdx.z;
    const int ty0  = blockIdx.y * TILE, tx0 = blockIdx.x * TILE;

    const float* img  = est_in + (size_t)b * IMH * IMW;
    const float* uimg = usig   + (size_t)b * IMH * IMW;
    const float* nimg = noisy  + (size_t)b * IMH * IMW;
    float*       oimg = est_out + (size_t)b * IMH * IMW;

    const float inv_umax = 1.f / fmaxf(__uint_as_float(umax[b]), EPSV);
    const float lam = lamp[0];
    const float gg  = 1.f / (1.f + __expf(-gatep[0]));   // sigmoid(gate)

    // ---- Phase 0: est tile via async global->LDS (reflect gather); iw tile scaled in VALU ----
    for (int i = tid; i < 36 * 36; i += 256) {
        int ly = i / 36, lx = i % 36;
        int gy = reflect_idx(ty0 + ly - 2, IMH);
        int gx = reflect_idx(tx0 + lx - 2, IMW);
        async_g2lds_b32(img + gy * IMW + gx, &sEst[i]);
    }
    for (int i = tid; i < 34 * 34; i += 256) {
        int ly = i / 34, lx = i % 34;
        int gy = min(max(ty0 + ly - 1, 0), IMH - 1);   // out-of-image values never used
        int gx = min(max(tx0 + lx - 1, 0), IMW - 1);
        sIw[i] = fmaxf(uimg[gy * IMW + gx] * inv_umax, EPSV);
    }

    // A operands (uniform filter data, ISA 16-bit A layout).
    // A1[m=out-ch][k=tap]: lanes 0-15 carry K=0..7 in halves 0..7; lanes 16-31 carry K=8 in half 0.
    // A2[m=tap][k=ch]   : tmp[t,p] = sum_c filt[c][t] * z[c,p]; K=8..15 rows are zero.
    v16h A1 = {}; v16h A2 = {};
    if (lane < 16) {
        if (lane < NF) {
#pragma unroll
            for (int k = 0; k < 8; ++k) A1[k] = (_Float16)filt[lane * 9 + k];
        }
        if (lane < 9) {
#pragma unroll
            for (int c = 0; c < NF; ++c) A2[c] = (_Float16)filt[c * 9 + lane];
        }
    } else {
        if (lane - 16 < NF) A1[0] = (_Float16)filt[(lane - 16) * 9 + 8];
    }
    float wr[NF], br[NF];
#pragma unroll
    for (int r = 0; r < NF; ++r) { wr[r] = wv[r]; br[r] = bv[r]; }

    wait_async();
    __syncthreads();

    // ---- Phase 1: 34x34 z-region in 16-pixel groups; chained WMMAs, branchless masking ----
    for (int grp = wave; grp < 73; grp += 8) {
        int pix = grp * 16 + (lane & 15);
        bool pvalid = pix < 34 * 34;
        int p  = pvalid ? pix : 0;
        int py = p / 34, px = p % 34;
        int base = (py + 1) * 36 + (px + 1);   // center in sEst

        // B1: im2col column (K=taps). Lanes 0-15: taps 0..7; lanes 16-31: tap 8.
        v16h B1 = {};
        if (lane < 16) {
#pragma unroll
            for (int k = 0; k < 8; ++k) {
                int dy = k / 3 - 1, dx = k % 3 - 1;
                B1[k] = (_Float16)sEst[base + dy * 36 + dx];
            }
        } else {
            B1[0] = (_Float16)sEst[base + 36 + 1];   // tap 8: dy=+1,dx=+1
        }

        v8f c1 = {};
        c1 = __builtin_amdgcn_wmma_f32_16x16x32_f16(false, A1, false, B1,
                                                    (short)0, c1, false, false);

        // z[c] = iw * tanh(resp*w + b); zero outside image (transposed conv zero-pads).
        // Branchless: fold validity into the iw scale (cndmask instead of exec juggling).
        float iwv = sIw[py * 34 + px];
        int iy = ty0 + py - 1, ix = tx0 + px - 1;
        bool inimg = (iy >= 0) && (iy < IMH) && (ix >= 0) && (ix < IMW);
        float zscale = (pvalid && inimg && (lane < 16)) ? iwv : 0.f;

        v16h B2;
#pragma unroll
        for (int r = 0; r < NF; ++r)
            B2[r] = (_Float16)(zscale * fast_tanh(fmaf(c1[r], wr[r], br[r])));
#pragma unroll
        for (int r = NF; r < 16; ++r) B2[r] = (_Float16)0.f;

        v8f c2 = {};
        c2 = __builtin_amdgcn_wmma_f32_16x16x32_f16(false, A2, false, B2,
                                                    (short)0, c2, false, false);

        if (pvalid) {
            int o = py * 36 + px;
            if (lane < 16) {
#pragma unroll
                for (int r = 0; r < 8; ++r) sTmp[r * (34 * 36) + o] = c2[r];  // taps 0..7
            } else {
                sTmp[8 * (34 * 36) + o] = c2[0];                              // tap 8 (M=8)
            }
        }
    }
    __syncthreads();

    // ---- Phase 2: diff gather + fidelity + gate + clip; float4 per thread ----
    {
        int i  = tid;                    // 256 threads x 4 pixels = 32x32 tile
        int oy = i >> 3;                 // 0..31
        int ox4 = (i & 7) * 4;           // 0,4,...,28
        float4 nv = *(const float4*)(nimg + (size_t)(ty0 + oy) * IMW + tx0 + ox4);
        float r[4];
#pragma unroll
        for (int j = 0; j < 4; ++j) {
            int ox = ox4 + j;
            float diff = 0.f;
#pragma unroll
            for (int t = 0; t < 9; ++t) {
                int dy = t / 3 - 1, dx = t % 3 - 1;
                diff += sTmp[t * (34 * 36) + (oy + 1 + dy) * 36 + (ox + 1 + dx)];
            }
            float e   = sEst[(oy + 2) * 36 + (ox + 2)];
            float nvx = (j == 0) ? nv.x : (j == 1) ? nv.y : (j == 2) ? nv.z : nv.w;
            float iwv = sIw[(oy + 1) * 34 + (ox + 1)];
            float fid = (e - nvx) / (e * e + EPSV);
            float est2 = e - diff - lam * iwv * fid;
            float outv = est2 + gg * (nvx - est2);
            r[j] = fminf(fmaxf(outv, EPSV), 1.f);
        }
        float4 o4 = make_float4(r[0], r[1], r[2], r[3]);
        *(float4*)(oimg + (size_t)(ty0 + oy) * IMW + tx0 + ox4) = o4;
    }
}

extern "C" void kernel_launch(void* const* d_in, const int* in_sizes, int n_in,
                              void* d_out, int out_size, void* d_ws, size_t ws_size,
                              hipStream_t stream) {
    const float* noisy   = (const float*)d_in[0];
    const float* filters = (const float*)d_in[1];   // [7][8][1][3][3]
    const float* wv      = (const float*)d_in[2];   // [7][8]
    const float* bv      = (const float*)d_in[3];   // [7][8]
    const float* lam     = (const float*)d_in[4];   // [7]
    const float* gate    = (const float*)d_in[5];   // [7]
    const float* gauss   = (const float*)d_in[6];   // [1][1][5][5]
    // d_in[7] = looks, unused
    float* out = (float*)d_out;

    const size_t IMG_BYTES = (size_t)8 * IMH * IMW * sizeof(float);
    char* ws = (char*)d_ws;
    float*        usig = (float*)(ws);
    float*        estA = (float*)(ws + IMG_BYTES);
    float*        estB = (float*)(ws + 2 * IMG_BYTES);
    unsigned int* umax = (unsigned int*)(ws + 3 * IMG_BYTES);

    init_umax_kernel<<<1, 256, 0, stream>>>(umax);   // atomicMax targets must start at 0 every call

    dim3 grid(IMW / TILE, IMH / TILE, 8);
    dim3 blk(256);
    for (int s = 0; s < NSTAGES; ++s) {
        const float* in  = (s == 0) ? noisy : ((s & 1) ? estA : estB);
        float*       ob  = (s == NSTAGES - 1) ? out : ((s & 1) ? estB : estA);
        gauss_max_kernel<<<grid, blk, 0, stream>>>(in, gauss, usig, umax + s * 8);
        stage_update_kernel<<<grid, blk, 0, stream>>>(in, noisy, usig, umax + s * 8,
                                                      filters + s * 72, wv + s * 8, bv + s * 8,
                                                      lam + s, gate + s, ob);
    }
}
